// JiT_90769838833770
// MI455X (gfx1250) — compile-verified, matrix operand order checked by reading, not run on hardware
//
#include <hip/hip_runtime.h>
#include <hip/hip_bf16.h>

typedef __attribute__((ext_vector_type(16))) _Float16 v16h;
typedef __attribute__((ext_vector_type(8)))  _Float16 v8h;
typedef __attribute__((ext_vector_type(8)))  float    v8f;

#define DD   768
#define NTOK 256
#define BB   16
#define HH   12
#define HD   64
#define LL   12
#define MTOT (BB*NTOK)      // 4096
#define INNER 2048

// ---------------------------------------------------------------- helpers
__device__ __forceinline__ v8f vzero8() {
    v8f z;
#pragma unroll
    for (int i = 0; i < 8; i++) z[i] = 0.f;
    return z;
}

// Build a 16-bit A/B WMMA fragment from a contiguous row of halves:
// lane holds K = kb..kb+7 and kb+16..kb+23. Two 16-byte vector loads so the
// backend emits ds_load_b128 / global_load_b128.
__device__ __forceinline__ v16h load_frag(const _Float16* p, int kb) {
    v8h lo = *(const v8h*)(p + kb);
    v8h hi = *(const v8h*)(p + kb + 16);
    v16h r;
#pragma unroll
    for (int i = 0; i < 8; i++) { r[i] = lo[i]; r[8 + i] = hi[i]; }
    return r;
}

__device__ __forceinline__ float sigmoidf_(float x) { return 1.f / (1.f + __expf(-x)); }

// ---------------------------------------------------------------- GEMM
// C[M,N] = A[M,K] (f32) @ W[K,N] (f32) + bias, optional fused epilogue:
//   resid != nullptr:  C = resid + gate[b, col] * (A@W + bias),  b = row>>8
// Block: 256 threads (8 waves), tile 128(M) x 64(N), K-step 32.
// Register-double-buffered staging: next tile's global loads are issued while
// WMMAs consume the current LDS tile.
#define BM 128
#define BN 64
#define BK 32
#define ALD 40
#define BLD 40

template <bool GUARD>
__global__ __launch_bounds__(256)
void gemm_wmma(const float* __restrict__ A, const float* __restrict__ W,
               const float* __restrict__ bias, float* __restrict__ C,
               const float* __restrict__ resid, const float* __restrict__ gate,
               int gate_ld, int M, int N, int K) {
    __shared__ _Float16 As[BM * ALD];
    __shared__ _Float16 Bs[BN * BLD];

    const int tid  = threadIdx.x;
    const int lane = tid & 31;
    const int wave = tid >> 5;
    const int wm   = wave >> 1;     // 0..3  (M direction)
    const int wn   = wave & 1;      // 0..1  (N direction)
    const int m0   = blockIdx.y * BM;
    const int n0   = blockIdx.x * BN;
    const int lr   = lane & 15;
    const int kb   = (lane >> 4) * 8;

    // staging coordinates (fixed per thread)
    int arow[4], ac4[4];
#pragma unroll
    for (int j = 0; j < 4; j++) {
        int i = tid + j * 256;
        arow[j] = i >> 3;
        ac4[j]  = (i & 7) * 4;
    }
    int brow[2], bc4[2];
#pragma unroll
    for (int j = 0; j < 2; j++) {
        int i = tid + j * 256;
        brow[j] = i >> 4;
        bc4[j]  = (i & 15) * 4;
    }

    v8f acc[2][2];
#pragma unroll
    for (int i = 0; i < 2; i++)
#pragma unroll
        for (int j = 0; j < 2; j++) acc[i][j] = vzero8();

    float4 va[4], vb[2];
    // prefetch first tile into registers
#pragma unroll
    for (int j = 0; j < 4; j++) {
        if (GUARD) {
            va[j] = make_float4(0.f, 0.f, 0.f, 0.f);
            if (m0 + arow[j] < M)
                va[j] = *(const float4*)(A + (size_t)(m0 + arow[j]) * K + ac4[j]);
        } else {
            va[j] = *(const float4*)(A + (size_t)(m0 + arow[j]) * K + ac4[j]);
        }
    }
#pragma unroll
    for (int j = 0; j < 2; j++)
        vb[j] = *(const float4*)(W + (size_t)brow[j] * N + n0 + bc4[j]);

    for (int k0 = 0; k0 < K; k0 += BK) {
        // store staged registers to LDS (f32 -> f16)
#pragma unroll
        for (int j = 0; j < 4; j++) {
            _Float16* d = &As[arow[j] * ALD + ac4[j]];
            d[0] = (_Float16)va[j].x; d[1] = (_Float16)va[j].y;
            d[2] = (_Float16)va[j].z; d[3] = (_Float16)va[j].w;
        }
#pragma unroll
        for (int j = 0; j < 2; j++) {
            Bs[(bc4[j] + 0) * BLD + brow[j]] = (_Float16)vb[j].x;
            Bs[(bc4[j] + 1) * BLD + brow[j]] = (_Float16)vb[j].y;
            Bs[(bc4[j] + 2) * BLD + brow[j]] = (_Float16)vb[j].z;
            Bs[(bc4[j] + 3) * BLD + brow[j]] = (_Float16)vb[j].w;
        }
        __syncthreads();

        // issue next tile's global loads while this tile is consumed
        const int kn = k0 + BK;
        if (kn < K) {
#pragma unroll
            for (int j = 0; j < 4; j++) {
                if (GUARD) {
                    va[j] = make_float4(0.f, 0.f, 0.f, 0.f);
                    if (m0 + arow[j] < M)
                        va[j] = *(const float4*)(A + (size_t)(m0 + arow[j]) * K + kn + ac4[j]);
                } else {
                    va[j] = *(const float4*)(A + (size_t)(m0 + arow[j]) * K + kn + ac4[j]);
                }
            }
#pragma unroll
            for (int j = 0; j < 2; j++)
                vb[j] = *(const float4*)(W + (size_t)(kn + brow[j]) * N + n0 + bc4[j]);
            // pull the tile after next toward L2 (speculative prefetch)
            if (kn + BK < K) {
                if (!GUARD || (m0 + arow[0]) < M)
                    __builtin_prefetch(A + (size_t)(m0 + arow[0]) * K + kn + BK, 0, 1);
                __builtin_prefetch(W + (size_t)(kn + BK + brow[0]) * N + n0, 0, 1);
            }
        }

        // fragments + WMMA
        v16h af[2], bf[2];
#pragma unroll
        for (int t = 0; t < 2; t++) {
            af[t] = load_frag(&As[(wm * 32 + t * 16 + lr) * ALD], kb);
            bf[t] = load_frag(&Bs[(wn * 32 + t * 16 + lr) * BLD], kb);
        }
#pragma unroll
        for (int tm = 0; tm < 2; tm++)
#pragma unroll
            for (int tn = 0; tn < 2; tn++)
                acc[tm][tn] = __builtin_amdgcn_wmma_f32_16x16x32_f16(
                    false, af[tm], false, bf[tn], (short)0, acc[tm][tn], false, false);
        __syncthreads();
    }

    // epilogue: lane = column lr within tile; rows m = (lane>>4)*8 + r
#pragma unroll
    for (int tm = 0; tm < 2; tm++)
#pragma unroll
        for (int tn = 0; tn < 2; tn++) {
            int col = n0 + wn * 32 + tn * 16 + lr;
            float bv = bias ? bias[col] : 0.f;
#pragma unroll
            for (int r = 0; r < 8; r++) {
                int m = m0 + wm * 32 + tm * 16 + (lane >> 4) * 8 + r;
                if (!GUARD || m < M) {
                    float v = acc[tm][tn][r] + bv;
                    if (resid) {
                        int b = m >> 8;
                        v = resid[(size_t)m * N + col] +
                            gate[(size_t)b * gate_ld + col] * v;
                    }
                    C[(size_t)m * N + col] = v;
                }
            }
        }
}

// ---------------------------------------------------------------- conditioning
__global__ __launch_bounds__(256)
void cond_kernel(const float* __restrict__ t, const int* __restrict__ y,
                 const float* __restrict__ Wt1, const float* __restrict__ bt1,
                 const float* __restrict__ Wt2, const float* __restrict__ bt2,
                 const float* __restrict__ emb_y, float* __restrict__ c_silu) {
    int b = blockIdx.x;
    __shared__ float tf[256];
    __shared__ float hid[DD];
    float tb = t[b];
    int i = threadIdx.x;
    {
        int fi = i & 127;
        float fr = __expf(-logf(10000.f) * (float)fi / 128.f);
        float a  = tb * fr;
        tf[i] = (i < 128) ? cosf(a) : sinf(a);
    }
    __syncthreads();
    for (int o = i; o < DD; o += 256) {
        float s = bt1[o];
        for (int k = 0; k < 256; k++) s += tf[k] * Wt1[(size_t)k * DD + o];
        hid[o] = s * sigmoidf_(s);
    }
    __syncthreads();
    int yb = y[b];
    for (int o = i; o < DD; o += 256) {
        float s = bt2[o];
        for (int k = 0; k < DD; k++) s += hid[k] * Wt2[(size_t)k * DD + o];
        s += emb_y[(size_t)yb * DD + o];
        c_silu[(size_t)b * DD + o] = s * sigmoidf_(s);
    }
}

// ---------------------------------------------------------------- patchify / unpatchify
__global__ __launch_bounds__(256)
void patchify_kernel(const float* __restrict__ x, float* __restrict__ xp) {
    for (size_t idx = blockIdx.x * 256 + threadIdx.x; idx < (size_t)MTOT * DD;
         idx += (size_t)gridDim.x * 256) {
        int q  = idx % DD;
        int n  = (idx / DD) % NTOK;
        int b  = idx / ((size_t)DD * NTOK);
        int cc = q % 3, pxy = q / 3;
        int px = pxy % 16, py = pxy / 16;
        int gx = n % 16, gy = n / 16;
        xp[idx] = x[(((size_t)b * 256 + gy * 16 + py) * 256 + gx * 16 + px) * 3 + cc];
    }
}

__global__ __launch_bounds__(256)
void unpatchify_kernel(const float* __restrict__ po, float* __restrict__ out) {
    for (size_t idx = blockIdx.x * 256 + threadIdx.x; idx < (size_t)MTOT * DD;
         idx += (size_t)gridDim.x * 256) {
        int q  = idx % DD;
        int n  = (idx / DD) % NTOK;
        int b  = idx / ((size_t)DD * NTOK);
        int cc = q % 3, pxy = q / 3;
        int px = pxy % 16, py = pxy / 16;
        int gx = n % 16, gy = n / 16;
        out[(((size_t)b * 256 + gy * 16 + py) * 256 + gx * 16 + px) * 3 + cc] = po[idx];
    }
}

// ---------------------------------------------------------------- rmsnorm + adaLN modulation
__global__ __launch_bounds__(256)
void modnorm_kernel(const float* __restrict__ x, const float* __restrict__ g,
                    const float* __restrict__ mod, int shOff, int scOff, int modld,
                    float* __restrict__ out) {
    int row = blockIdx.x;
    int b   = row >> 8;
    __shared__ float red[8];
    float ssq = 0.f;
    for (int c = threadIdx.x; c < DD; c += 256) {
        float v = x[(size_t)row * DD + c];
        ssq += v * v;
    }
#pragma unroll
    for (int o = 16; o >= 1; o >>= 1) ssq += __shfl_xor(ssq, o, 32);
    if ((threadIdx.x & 31) == 0) red[threadIdx.x >> 5] = ssq;
    __syncthreads();
    float tot = 0.f;
#pragma unroll
    for (int i = 0; i < 8; i++) tot += red[i];
    float r = rsqrtf(tot / (float)DD + 1e-6f);
    for (int c = threadIdx.x; c < DD; c += 256) {
        float v = x[(size_t)row * DD + c] * r * g[c];
        out[(size_t)row * DD + c] =
            v * (1.f + mod[(size_t)b * modld + scOff + c]) + mod[(size_t)b * modld + shOff + c];
    }
}

// ---------------------------------------------------------------- q/k rmsnorm + rope, v transpose
// one wave per (b, n, h); lane handles d=lane and d=lane+32
__global__ __launch_bounds__(256)
void rope_rms_kernel(const float* __restrict__ qkv, const float* __restrict__ gq,
                     const float* __restrict__ gk, _Float16* __restrict__ qr,
                     _Float16* __restrict__ kr, _Float16* __restrict__ vT) {
    int gw = blockIdx.x * 8 + (threadIdx.x >> 5);
    if (gw >= BB * NTOK * HH) return;
    int lane = threadIdx.x & 31;
    int h = gw % HH;
    int n = (gw / HH) % NTOK;
    int b = gw / (HH * NTOK);
    const size_t rowbase = ((size_t)(b * NTOK + n)) * (3 * DD) + (size_t)h * HD;

    // rope angles for d0=lane, d1=lane+32
    int fi = lane >> 1;
    float freq = __expf(-(float)fi * (logf(10000.f) / 16.f));
    float a0 = (float)(n >> 4) * freq;   // gy for d < 32
    float a1 = (float)(n & 15) * freq;   // gx for d >= 32
    float cs0 = cosf(a0), sn0 = sinf(a0);
    float cs1 = cosf(a1), sn1 = sinf(a1);
    float sgn = (lane & 1) ? 1.f : -1.f;

    const size_t bh  = (size_t)b * HH + h;
    const size_t ob  = (bh * NTOK + n) * HD;

#pragma unroll
    for (int s = 0; s < 2; s++) {  // s=0: q (gq), s=1: k (gk)
        const float* src = qkv + rowbase + (size_t)s * DD;
        const float* gg  = s ? gk : gq;
        float x0 = src[lane], x1 = src[lane + 32];
        float ssq = x0 * x0 + x1 * x1;
#pragma unroll
        for (int o = 16; o >= 1; o >>= 1) ssq += __shfl_xor(ssq, o, 32);
        float r = rsqrtf(ssq / (float)HD + 1e-6f);
        x0 *= r * gg[lane];
        x1 *= r * gg[lane + 32];
        float p0 = __shfl_xor(x0, 1, 32);
        float p1 = __shfl_xor(x1, 1, 32);
        float o0 = x0 * cs0 + sgn * p0 * sn0;
        float o1 = x1 * cs1 + sgn * p1 * sn1;
        _Float16* dst = s ? kr : qr;
        dst[ob + lane]      = (_Float16)o0;
        dst[ob + lane + 32] = (_Float16)o1;
    }
    // v: transpose into [b][h][d][n]
    const float* vs = qkv + rowbase + 2 * DD;
    vT[(bh * HD + lane) * NTOK + n]        = (_Float16)vs[lane];
    vT[(bh * HD + lane + 32) * NTOK + n]   = (_Float16)vs[lane + 32];
}

// ---------------------------------------------------------------- attention
// grid.x = B*H, grid.y = 4 (64 query rows); block = 128 (4 waves, 16 q-rows each)
__global__ __launch_bounds__(128)
void attn_kernel(const _Float16* __restrict__ qr, const _Float16* __restrict__ kr,
                 const _Float16* __restrict__ vT, float* __restrict__ O) {
    __shared__ _Float16 S[4][16][264];
    const int bh   = blockIdx.x;
    const int b    = bh / HH, h = bh % HH;
    const int wave = threadIdx.x >> 5;
    const int lane = threadIdx.x & 31;
    const int lr   = lane & 15;
    const int kb   = (lane >> 4) * 8;
    const int q0   = blockIdx.y * 64 + wave * 16;
    const size_t base = (size_t)bh * NTOK * HD;

    // Q fragments (two K-steps over HD=64)
    v16h aq[2];
#pragma unroll
    for (int s = 0; s < 2; s++)
        aq[s] = load_frag(qr + base + (size_t)(q0 + lr) * HD + s * 32, kb);

    const float scale = 0.125f;  // 1/sqrt(64)
    for (int kt = 0; kt < 16; kt++) {
        v8f acc = vzero8();
#pragma unroll
        for (int s = 0; s < 2; s++) {
            v16h bk = load_frag(kr + base + (size_t)(kt * 16 + lr) * HD + s * 32, kb);
            acc = __builtin_amdgcn_wmma_f32_16x16x32_f16(false, aq[s], false, bk,
                                                         (short)0, acc, false, false);
        }
#pragma unroll
        for (int r = 0; r < 8; r++) {
            int m = (lane >> 4) * 8 + r;
            S[wave][m][kt * 16 + lr] = (_Float16)(acc[r] * scale);
        }
    }

    // softmax over 256 cols; lane pair (l, l+16) splits row lr
    {
        int c0 = (lane >> 4) * 128;
        float mx = -1e30f;
        for (int c = 0; c < 128; c++) mx = fmaxf(mx, (float)S[wave][lr][c0 + c]);
        mx = fmaxf(mx, __shfl_xor(mx, 16, 32));
        float sum = 0.f;
        for (int c = 0; c < 128; c++) {
            float e = __expf((float)S[wave][lr][c0 + c] - mx);
            S[wave][lr][c0 + c] = (_Float16)e;
            sum += e;
        }
        sum += __shfl_xor(sum, 16, 32);
        float inv = 1.f / sum;
        for (int c = 0; c < 128; c++)
            S[wave][lr][c0 + c] = (_Float16)((float)S[wave][lr][c0 + c] * inv);
    }

    // O = P @ V : dt tiles over HD, ks steps over keys
    for (int dt = 0; dt < 4; dt++) {
        v8f acc = vzero8();
#pragma unroll
        for (int ks = 0; ks < 8; ks++) {
            v16h ap = load_frag(&S[wave][lr][ks * 32], kb);
            v16h bv = load_frag(vT + ((size_t)bh * HD + dt * 16 + lr) * NTOK + ks * 32, kb);
            acc = __builtin_amdgcn_wmma_f32_16x16x32_f16(false, ap, false, bv,
                                                         (short)0, acc, false, false);
        }
#pragma unroll
        for (int r = 0; r < 8; r++) {
            int m = (lane >> 4) * 8 + r;   // query row within tile
            int n = q0 + m;
            O[((size_t)(b * NTOK + n)) * DD + h * HD + dt * 16 + lr] = acc[r];
        }
    }
}

// ---------------------------------------------------------------- swiglu
__global__ __launch_bounds__(256)
void swiglu_kernel(const float* __restrict__ x12, float* __restrict__ xg) {
    for (size_t idx = blockIdx.x * 256 + threadIdx.x; idx < (size_t)MTOT * INNER;
         idx += (size_t)gridDim.x * 256) {
        size_t row = idx / INNER;
        int    c   = idx % INNER;
        float x1 = x12[row * (2 * INNER) + c];
        float x2 = x12[row * (2 * INNER) + INNER + c];
        xg[idx] = x1 * sigmoidf_(x1) * x2;
    }
}

// ---------------------------------------------------------------- launch
extern "C" void kernel_launch(void* const* d_in, const int* in_sizes, int n_in,
                              void* d_out, int out_size, void* d_ws, size_t ws_size,
                              hipStream_t stream) {
    const float* x     = (const float*)d_in[0];
    const float* t     = (const float*)d_in[1];
    const int*   y     = (const int*)d_in[2];
    const float* W_pe1 = (const float*)d_in[3];
    const float* W_pe2 = (const float*)d_in[4];
    const float* b_pe2 = (const float*)d_in[5];
    const float* Wt1   = (const float*)d_in[6];
    const float* bt1   = (const float*)d_in[7];
    const float* Wt2   = (const float*)d_in[8];
    const float* bt2   = (const float*)d_in[9];
    const float* emb_y = (const float*)d_in[10];
    const float* g1    = (const float*)d_in[11];
    const float* Wqkv  = (const float*)d_in[12];
    const float* bqkv  = (const float*)d_in[13];
    const float* gq    = (const float*)d_in[14];
    const float* gk    = (const float*)d_in[15];
    const float* Wo    = (const float*)d_in[16];
    const float* bo    = (const float*)d_in[17];
    const float* g2    = (const float*)d_in[18];
    const float* W12   = (const float*)d_in[19];
    const float* b12   = (const float*)d_in[20];
    const float* W3    = (const float*)d_in[21];
    const float* b3    = (const float*)d_in[22];
    const float* Wada  = (const float*)d_in[23];
    const float* bada  = (const float*)d_in[24];
    const float* gF    = (const float*)d_in[25];
    const float* WadaF = (const float*)d_in[26];
    const float* badaF = (const float*)d_in[27];
    const float* Wout  = (const float*)d_in[28];
    const float* bout  = (const float*)d_in[29];

    float* ws = (float*)d_ws;
    size_t o = 0;
    float* c_silu = ws + o; o += (size_t)BB * DD;
    float* xp     = ws + o; o += (size_t)MTOT * DD;
    float* petmp  = ws + o; o += (size_t)MTOT * 128;
    float* xc     = ws + o; o += (size_t)MTOT * DD;
    float* mod    = ws + o; o += (size_t)BB * 6 * DD;
    float* hbuf   = ws + o; o += (size_t)MTOT * DD;
    float* qkv    = ws + o; o += (size_t)MTOT * 3 * DD;
    float* attO   = ws + o; o += (size_t)MTOT * DD;
    float* x12    = ws + o; o += (size_t)MTOT * 2 * INNER;
    float* xg     = ws + o; o += (size_t)MTOT * INNER;
    float* modf   = ws + o; o += (size_t)BB * 2 * DD;
    _Float16* qr = (_Float16*)(ws + o); o += (size_t)BB * HH * NTOK * HD / 2;
    _Float16* kr = (_Float16*)(ws + o); o += (size_t)BB * HH * NTOK * HD / 2;
    _Float16* vT = (_Float16*)(ws + o); o += (size_t)BB * HH * NTOK * HD / 2;
    float* po = xp;  // reuse patch buffer at the end

    // conditioning
    cond_kernel<<<BB, 256, 0, stream>>>(t, y, Wt1, bt1, Wt2, bt2, emb_y, c_silu);

    // patch embed
    patchify_kernel<<<12288, 256, 0, stream>>>(x, xp);
    gemm_wmma<false><<<dim3(128 / 64, MTOT / 128), 256, 0, stream>>>(
        xp, W_pe1, nullptr, petmp, nullptr, nullptr, 0, MTOT, 128, DD);
    gemm_wmma<false><<<dim3(DD / 64, MTOT / 128), 256, 0, stream>>>(
        petmp, W_pe2, b_pe2, xc, nullptr, nullptr, 0, MTOT, DD, 128);

    for (int l = 0; l < LL; l++) {
        const float* Wqkv_l = Wqkv + (size_t)l * DD * 3 * DD;
        const float* bqkv_l = bqkv + (size_t)l * 3 * DD;
        const float* Wo_l   = Wo   + (size_t)l * DD * DD;
        const float* bo_l   = bo   + (size_t)l * DD;
        const float* W12_l  = W12  + (size_t)l * DD * 2 * INNER;
        const float* b12_l  = b12  + (size_t)l * 2 * INNER;
        const float* W3_l   = W3   + (size_t)l * INNER * DD;
        const float* b3_l   = b3   + (size_t)l * DD;
        const float* Wada_l = Wada + (size_t)l * DD * 6 * DD;
        const float* bada_l = bada + (size_t)l * 6 * DD;

        // mod = c_silu @ Wada + bada   (16 x 4608)
        gemm_wmma<true><<<dim3(6 * DD / 64, 1), 256, 0, stream>>>(
            c_silu, Wada_l, bada_l, mod, nullptr, nullptr, 0, BB, 6 * DD, DD);

        // h = rms(xc, g1)*(1+sc_a) + sh_a
        modnorm_kernel<<<MTOT, 256, 0, stream>>>(xc, g1 + l * DD, mod, 0, DD, 6 * DD, hbuf);

        // qkv
        gemm_wmma<false><<<dim3(3 * DD / 64, MTOT / 128), 256, 0, stream>>>(
            hbuf, Wqkv_l, bqkv_l, qkv, nullptr, nullptr, 0, MTOT, 3 * DD, DD);

        // q/k rmsnorm + rope, v transpose
        rope_rms_kernel<<<(BB * NTOK * HH) / 8, 256, 0, stream>>>(
            qkv, gq + l * HD, gk + l * HD, qr, kr, vT);

        // attention
        attn_kernel<<<dim3(BB * HH, 4), 128, 0, stream>>>(qr, kr, vT, attO);

        // xc = xc + g_a * (attO @ Wo + bo)
        gemm_wmma<false><<<dim3(DD / 64, MTOT / 128), 256, 0, stream>>>(
            attO, Wo_l, bo_l, xc, xc, mod + 2 * DD, 6 * DD, MTOT, DD, DD);

        // h2 = rms(xc, g2)*(1+sc_m) + sh_m
        modnorm_kernel<<<MTOT, 256, 0, stream>>>(xc, g2 + l * DD, mod, 3 * DD, 4 * DD, 6 * DD, hbuf);

        // x12 = h2 @ W12 + b12
        gemm_wmma<false><<<dim3(2 * INNER / 64, MTOT / 128), 256, 0, stream>>>(
            hbuf, W12_l, b12_l, x12, nullptr, nullptr, 0, MTOT, 2 * INNER, DD);

        swiglu_kernel<<<32768, 256, 0, stream>>>(x12, xg);

        // xc = xc + g_m * (xg @ W3 + b3)
        gemm_wmma<false><<<dim3(DD / 64, MTOT / 128), 256, 0, stream>>>(
            xg, W3_l, b3_l, xc, xc, mod + 5 * DD, 6 * DD, MTOT, DD, INNER);
    }

    // final modulation + projection
    gemm_wmma<true><<<dim3(2 * DD / 64, 1), 256, 0, stream>>>(
        c_silu, WadaF, badaF, modf, nullptr, nullptr, 0, BB, 2 * DD, DD);
    modnorm_kernel<<<MTOT, 256, 0, stream>>>(xc, gF, modf, 0, DD, 2 * DD, hbuf);
    gemm_wmma<false><<<dim3(DD / 64, MTOT / 128), 256, 0, stream>>>(
        hbuf, Wout, bout, po, nullptr, nullptr, 0, MTOT, DD, DD);
    unpatchify_kernel<<<12288, 256, 0, stream>>>(po, (float*)d_out);
}
